// MessageBlock_48825188221159
// MI455X (gfx1250) — compile-verified, matrix operand order checked by reading, not run on hardware
//
#include <hip/hip_runtime.h>
#include <hip/hip_bf16.h>
#include <math.h>

typedef __attribute__((ext_vector_type(16))) _Float16 v16h;
typedef __attribute__((ext_vector_type(8)))  _Float16 v8h;
typedef __attribute__((ext_vector_type(8)))  float    v8f;

#define F_DIM 128

// B-fragment fetch: packed as frag-major [kt][nt][lane][16 halves], 32B/lane contiguous.
__device__ __forceinline__ v16h load_bfrag(const _Float16* __restrict__ pack, int kt, int nt, int lane) {
    const v16h* p = reinterpret_cast<const v16h*>(pack) + ((kt * 8 + nt) * 32 + lane);
    return *p;
}

// Fast SiLU: x * v_rcp_f32(1 + e^-x)  (avoids IEEE-div Newton expansion)
__device__ __forceinline__ float silu_f(float x) {
    return x * __builtin_amdgcn_rcpf(1.0f + __expf(-x));
}

// Wave-level 16-row MLP: D16x128 = silu(A @ W1 + b1) @ W2 + b2, emitted per 16-col tile.
// A frags follow the 16-bit 16x32 A layout (ISA 7.12.2); H staged through LDS to
// re-shape the f32 D-layout into layer-2 A frags.
template<int KT1, class EmitFn>
__device__ __forceinline__ void run_mlp(const v16h* A,
                                        const _Float16* __restrict__ W1p, const float* __restrict__ B1,
                                        const _Float16* __restrict__ W2p, const float* __restrict__ B2,
                                        _Float16* Hbase, int lane, EmitFn emit) {
    const int hi   = lane >> 4;
    const int nloc = lane & 15;
    // hoist bias loads so they clause together and retire before the WMMAs
    float b1v[8], b2v[8];
#pragma unroll
    for (int nt = 0; nt < 8; ++nt) b1v[nt] = B1[nt * 16 + nloc];
#pragma unroll
    for (int nt = 0; nt < 8; ++nt) b2v[nt] = B2[nt * 16 + nloc];

    // ---- layer 1 + SiLU -> stage H[16x128] f16 into LDS
    for (int nt = 0; nt < 8; ++nt) {
        float b = b1v[nt];
        v8f acc;
#pragma unroll
        for (int i = 0; i < 8; ++i) acc[i] = b;
#pragma unroll
        for (int kt = 0; kt < KT1; ++kt) {
            v16h bf = load_bfrag(W1p, kt, nt, lane);
            acc = __builtin_amdgcn_wmma_f32_16x16x32_f16(false, A[kt], false, bf,
                                                         (short)0, acc, false, false);
        }
#pragma unroll
        for (int r = 0; r < 8; ++r) {
            // D layout: VGPR r -> (m = r + 8*hi, n = nt*16 + nloc)
            Hbase[(r + hi * 8) * F_DIM + nt * 16 + nloc] = (_Float16)silu_f(acc[r]);
        }
    }
    asm volatile("s_wait_dscnt 0x0" ::: "memory");
    // ---- layer 2 A frags from LDS (A layout: row m = nloc; K runs c0, c0+16)
    const _Float16* Hrow = Hbase + nloc * F_DIM;
    v16h A2[4];
#pragma unroll
    for (int kt = 0; kt < 4; ++kt) {
        int c0 = kt * 32 + hi * 8;
        v8h lo  = *reinterpret_cast<const v8h*>(Hrow + c0);
        v8h hi8 = *reinterpret_cast<const v8h*>(Hrow + c0 + 16);
#pragma unroll
        for (int j = 0; j < 8; ++j) { A2[kt][j] = lo[j]; A2[kt][j + 8] = hi8[j]; }
    }
    for (int nt = 0; nt < 8; ++nt) {
        float b = b2v[nt];
        v8f acc;
#pragma unroll
        for (int i = 0; i < 8; ++i) acc[i] = b;
#pragma unroll
        for (int kt = 0; kt < 4; ++kt) {
            v16h bf = load_bfrag(W2p, kt, nt, lane);
            acc = __builtin_amdgcn_wmma_f32_16x16x32_f16(false, A2[kt], false, bf,
                                                         (short)0, acc, false, false);
        }
        emit(nt, acc);
    }
    asm volatile("s_wait_dscnt 0x0" ::: "memory");
}

// Pre-swizzle a [Kin x 128] fp32 weight matrix (K-padded to Kp) into per-lane f16
// B-fragments: lane<16 -> (n = nt*16+lane, K = kt*32 + {0..7,16..23}); lane>=16 ->
// (n = nt*16+lane-16, K = kt*32 + {8..15,24..31}).
__global__ void pack_weights_kernel(const float* __restrict__ W, _Float16* __restrict__ dst,
                                    int Kin, int Kp) {
    int t = blockIdx.x * 256 + threadIdx.x;
    int total = (Kp / 32) * 8 * 512;
    if (t >= total) return;
    int within  = t & 511;
    int fragIdx = t >> 9;
    int lane = within >> 4;
    int h    = within & 15;
    int kt = fragIdx >> 3;
    int nt = fragIdx & 7;
    int hi   = lane >> 4;
    int nloc = lane & 15;
    int n = nt * 16 + nloc;
    int k = kt * 32 + hi * 8 + h + ((h >= 8) ? 8 : 0);
    float val = (k < Kin) ? W[(size_t)k * F_DIM + n] : 0.0f;
    dst[t] = (_Float16)val;
}

// One wave per 16-edge tile: both message MLPs (shared input frags), fused SiLU,
// fused v_diff normalization, f32 atomic scatter into segment sums.
__global__ void __launch_bounds__(256) edge_msg_kernel(
    const float* __restrict__ s, const float* __restrict__ v,
    const int* __restrict__ eidx, const float* __restrict__ rbf,
    const _Float16* __restrict__ msW1p, const float* __restrict__ msB1,
    const _Float16* __restrict__ msW2p, const float* __restrict__ msB2,
    const _Float16* __restrict__ mvW1p, const float* __restrict__ mvB1,
    const _Float16* __restrict__ mvW2p, const float* __restrict__ mvB2,
    float* __restrict__ dsAgg, float* __restrict__ dvAgg,
    int E, int nTiles)
{
    __shared__ _Float16 Hst[8 * 16 * F_DIM];
    const int wave = threadIdx.x >> 5;
    const int lane = threadIdx.x & 31;
    const int tile = blockIdx.x * 8 + wave;
    if (tile >= nTiles) return;                 // wave-uniform: EXEC stays all-ones
    const int e0   = tile * 16;
    const int hi   = lane >> 4;
    const int nloc = lane & 15;
    const int* __restrict__ rowp = eidx;
    const int* __restrict__ colp = eidx + E;

    int rows8[8], cols8[8];
#pragma unroll
    for (int r = 0; r < 8; ++r) {
        int e = e0 + r + hi * 8;
        rows8[r] = rowp[e];
        cols8[r] = colp[e];
    }

    // A frags for X = [s[row] | rbf | 0-pad] (160 cols), per the 16-bit A 16x32 layout.
    const int myRow = rowp[e0 + nloc];
    const float* __restrict__ srow = s + (size_t)myRow * F_DIM;
    const float* __restrict__ rrow = rbf + (size_t)(e0 + nloc) * 20;
    v16h A[5];
#pragma unroll
    for (int kt = 0; kt < 4; ++kt) {
        int c0 = kt * 32 + hi * 8;
        alignas(16) float buf[16];
        *(float4*)(&buf[0])  = *(const float4*)(srow + c0);
        *(float4*)(&buf[4])  = *(const float4*)(srow + c0 + 4);
        *(float4*)(&buf[8])  = *(const float4*)(srow + c0 + 16);
        *(float4*)(&buf[12]) = *(const float4*)(srow + c0 + 20);
#pragma unroll
        for (int j = 0; j < 16; ++j) A[kt][j] = (_Float16)buf[j];
    }
    {
        int c0 = 128 + hi * 8;
#pragma unroll
        for (int j = 0; j < 16; ++j) {
            int k = c0 + j + ((j >= 8) ? 8 : 0);
            A[4][j] = (_Float16)((k < 148) ? rrow[k - 128] : 0.0f);
        }
    }
    _Float16* Hbase = &Hst[wave * 16 * F_DIM];

    // scalar message MLP -> segment_sum(ds, col)
    run_mlp<5>(A, msW1p, msB1, msW2p, msB2, Hbase, lane,
        [&](int nt, v8f acc) {
            int n = nt * 16 + nloc;
#pragma unroll
            for (int r = 0; r < 8; ++r)
                atomicAdd(dsAgg + (size_t)cols8[r] * F_DIM + n, acc[r]);
        });

    // vector message MLP (same 3-axis-broadcast input) -> gate by v_diff/||v_diff|| -> segment_sum
    run_mlp<5>(A, mvW1p, mvB1, mvW2p, mvB2, Hbase, lane,
        [&](int nt, v8f acc) {
            int n = nt * 16 + nloc;
#pragma unroll
            for (int r = 0; r < 8; ++r) {
                const float* vr = v + (size_t)rows8[r] * (3 * F_DIM) + n;
                const float* vc = v + (size_t)cols8[r] * (3 * F_DIM) + n;
                float d0 = vr[0] - vc[0];
                float d1 = vr[F_DIM] - vc[F_DIM];
                float d2 = vr[2 * F_DIM] - vc[2 * F_DIM];
                float nrm = __builtin_amdgcn_sqrtf(fmaf(d0, d0, fmaf(d1, d1, d2 * d2)));
                float scl = acc[r] * __builtin_amdgcn_rcpf(nrm + 1e-8f);
                float* dst = dvAgg + (size_t)cols8[r] * (3 * F_DIM) + n;
                atomicAdd(dst,             scl * d0);
                atomicAdd(dst + F_DIM,     scl * d1);
                atomicAdd(dst + 2 * F_DIM, scl * d2);
            }
        });
}

// One wave per 16-row tile: out = Xa + MLP(concat(Xa, Xb)), K=256.
__global__ void __launch_bounds__(256) node_update_kernel(
    const float* __restrict__ Xa, const float* __restrict__ Xb,
    const _Float16* __restrict__ W1p, const float* __restrict__ B1,
    const _Float16* __restrict__ W2p, const float* __restrict__ B2,
    float* __restrict__ out, int nTiles)
{
    __shared__ _Float16 Hst[8 * 16 * F_DIM];
    const int wave = threadIdx.x >> 5;
    const int lane = threadIdx.x & 31;
    const int tile = blockIdx.x * 8 + wave;
    if (tile >= nTiles) return;
    const int t0   = tile * 16;
    const int hi   = lane >> 4;
    const int nloc = lane & 15;

    const float* __restrict__ xa = Xa + (size_t)(t0 + nloc) * F_DIM;
    const float* __restrict__ xb = Xb + (size_t)(t0 + nloc) * F_DIM;
    v16h A[8];
#pragma unroll
    for (int kt = 0; kt < 8; ++kt) {
        int c0 = kt * 32 + hi * 8;
        const float* p = (kt < 4) ? (xa + c0) : (xb + (c0 - 128));
        alignas(16) float buf[16];
        *(float4*)(&buf[0])  = *(const float4*)(p);
        *(float4*)(&buf[4])  = *(const float4*)(p + 4);
        *(float4*)(&buf[8])  = *(const float4*)(p + 16);
        *(float4*)(&buf[12]) = *(const float4*)(p + 20);
#pragma unroll
        for (int j = 0; j < 16; ++j) A[kt][j] = (_Float16)buf[j];
    }
    _Float16* Hbase = &Hst[wave * 16 * F_DIM];
    run_mlp<8>(A, W1p, B1, W2p, B2, Hbase, lane,
        [&](int nt, v8f acc) {
            int n = nt * 16 + nloc;
#pragma unroll
            for (int r = 0; r < 8; ++r) {
                size_t ri = (size_t)(t0 + r + hi * 8);
                out[ri * F_DIM + n] = Xa[ri * F_DIM + n] + acc[r];
            }
        });
}

extern "C" void kernel_launch(void* const* d_in, const int* in_sizes, int n_in,
                              void* d_out, int out_size, void* d_ws, size_t ws_size,
                              hipStream_t stream) {
    const float* s   = (const float*)d_in[0];
    const float* v   = (const float*)d_in[1];
    const int*   ei  = (const int*)d_in[2];
    const float* rbf = (const float*)d_in[3];
    const float* ms_w1 = (const float*)d_in[4];  const float* ms_b1 = (const float*)d_in[5];
    const float* ms_w2 = (const float*)d_in[6];  const float* ms_b2 = (const float*)d_in[7];
    const float* mv_w1 = (const float*)d_in[8];  const float* mv_b1 = (const float*)d_in[9];
    const float* mv_w2 = (const float*)d_in[10]; const float* mv_b2 = (const float*)d_in[11];
    const float* us_w1 = (const float*)d_in[12]; const float* us_b1 = (const float*)d_in[13];
    const float* us_w2 = (const float*)d_in[14]; const float* us_b2 = (const float*)d_in[15];
    const float* uv_w1 = (const float*)d_in[16]; const float* uv_b1 = (const float*)d_in[17];
    const float* uv_w2 = (const float*)d_in[18]; const float* uv_b2 = (const float*)d_in[19];

    const int N = in_sizes[0] / F_DIM;   // 10000
    const int E = in_sizes[2] / 2;       // 640000

    char* ws = (char*)d_ws;
    float* dsAgg = (float*)ws;                                           // [N,128] f32
    float* dvAgg = (float*)(ws + (size_t)N * F_DIM * sizeof(float));     // [N,3,128] f32
    size_t off = (size_t)N * F_DIM * sizeof(float) * 4;
    auto takePack = [&](int Kp) {
        _Float16* p = (_Float16*)(ws + off);
        off += (size_t)Kp * F_DIM * sizeof(_Float16);
        return p;
    };
    _Float16* msW1p = takePack(160);
    _Float16* msW2p = takePack(128);
    _Float16* mvW1p = takePack(160);
    _Float16* mvW2p = takePack(128);
    _Float16* usW1p = takePack(256);
    _Float16* usW2p = takePack(128);
    _Float16* uvW1p = takePack(256);
    _Float16* uvW2p = takePack(128);

    // zero the segment-sum accumulators (graph-capture-safe memset node)
    hipMemsetAsync(ws, 0, (size_t)N * F_DIM * sizeof(float) * 4, stream);

    auto pack = [&](const float* W, _Float16* dst, int Kin, int Kp) {
        int total = (Kp / 32) * 8 * 512;
        pack_weights_kernel<<<(total + 255) / 256, 256, 0, stream>>>(W, dst, Kin, Kp);
    };
    pack(ms_w1, msW1p, 148, 160);
    pack(ms_w2, msW2p, 128, 128);
    pack(mv_w1, mvW1p, 148, 160);
    pack(mv_w2, mvW2p, 128, 128);
    pack(us_w1, usW1p, 256, 256);
    pack(us_w2, usW2p, 128, 128);
    pack(uv_w1, uvW1p, 256, 256);
    pack(uv_w2, uvW2p, 128, 128);

    int eTiles = E / 16;
    edge_msg_kernel<<<(eTiles + 7) / 8, 256, 0, stream>>>(
        s, v, ei, rbf,
        msW1p, ms_b1, msW2p, ms_b2,
        mvW1p, mv_b1, mvW2p, mv_b2,
        dsAgg, dvAgg, E, eTiles);

    int sTiles = N / 16;
    node_update_kernel<<<(sTiles + 7) / 8, 256, 0, stream>>>(
        s, dsAgg, usW1p, us_b1, usW2p, us_b2, (float*)d_out, sTiles);

    int vTiles = (3 * N) / 16;
    node_update_kernel<<<(vTiles + 7) / 8, 256, 0, stream>>>(
        v, dvAgg, uvW1p, uv_b1, uvW2p, uv_b2, (float*)d_out + (size_t)N * F_DIM, vTiles);
}
